// NeuralCDE_35708358099523
// MI455X (gfx1250) — matrix operand order repaired and checked by reading, not compiled, not run on hardware
//
#include <hip/hip_runtime.h>
#include <hip/hip_bf16.h>
#include <math.h>

// Problem sizes
#define BN   64
#define SEQ  32
#define CN   16
#define HN   128
#define WN   256
#define OUTN 10
#define NI   31      // SEQ-1 intervals
#define NSTEP 62     // NI * K substeps

typedef __attribute__((ext_vector_type(16))) __bf16 v16bf;
typedef __attribute__((ext_vector_type(8)))  __bf16 v8bf;
typedef __attribute__((ext_vector_type(8)))  float  v8f;

// Tsit5 tableau (Tsitouras 2011)
__device__ __constant__ float TSA[6][5] = {
    {0.f, 0.f, 0.f, 0.f, 0.f},
    {0.161f, 0.f, 0.f, 0.f, 0.f},
    {-0.008480655492356989f, 0.335480655492357f, 0.f, 0.f, 0.f},
    {2.8971530571054935f, -6.359448489975075f, 4.3622954328695815f, 0.f, 0.f},
    {5.325864828439257f, -11.748883564062828f, 7.4955393428898365f, -0.09249506636175525f, 0.f},
    {5.86145544294642f, -12.92096931784711f, 8.159367898576159f, -0.071584973281401f, -0.028269050394068383f}};
__device__ __constant__ float TSC[6] = {0.f, 0.161f, 0.327f, 0.9f, 0.9800255409045097f, 1.f};
__device__ __constant__ float TSB[6] = {0.09646076681806523f, 0.01f, 0.4798896504144996f,
                                        1.379008574103742f, -3.290069515436081f, 2.324710524099774f};

__device__ inline __bf16 f2bf(float x) {
    union { __hip_bfloat16 h; __bf16 b; } u;
    u.h = __float2bfloat16(x);
    return u.b;
}
__device__ inline float softplusf(float x) {
    return fmaxf(x, 0.f) + logf(1.f + expf(-fabsf(x)));
}

// A fragment (16x32 bf16, M x K): lane l<16 holds row m=l, K = {kb..kb+15}
// with kb = kt*32; lanes 16..31 hold K = {kb+16..kb+31} (contiguous, matches B).
__device__ inline v16bf load_frag_a(const __bf16* __restrict__ base, int kt, int lane, int stride) {
    const int m  = lane & 15;
    const int kb = kt * 32 + ((lane >> 4) << 4);
    const __bf16* p = base + m * stride + kb;
    union { v16bf v; v8bf h[2]; } u;
    u.h[0] = *(const v8bf*)(p);
    u.h[1] = *(const v8bf*)(p + 8);
    return u.v;
}
// B fragment (32x16 bf16, K x N): lane l holds column n = nt*16 + (l&15),
// K = kt*32 + ((l>=16)?16:0) + e, e=0..15 -> 16 contiguous K of row-major w[n][k].
__device__ inline v16bf load_frag_b(const __bf16* __restrict__ base, int nt, int kt, int lane, int stride) {
    const int n  = nt * 16 + (lane & 15);
    const int kb = kt * 32 + ((lane >> 4) << 4);
    return *(const v16bf*)(base + n * stride + kb);
}

// ---------------- small helper kernels ----------------

__global__ void to_bf16_kernel(const float* __restrict__ src, __hip_bfloat16* __restrict__ dst, int n) {
    int i = blockIdx.x * blockDim.x + threadIdx.x;
    if (i < n) dst[i] = __float2bfloat16(src[i]);
}

// init MLP layer 0: x = coeff_a[:,0,:]  ->  relu(x @ iw0.T + ib0)   [64,256]
__global__ void init_l0(const float* __restrict__ ca, const float* __restrict__ iw0,
                        const float* __restrict__ ib0, float* __restrict__ h0) {
    int idx = blockIdx.x * blockDim.x + threadIdx.x;
    if (idx >= BN * WN) return;
    int b = idx / WN, w = idx % WN;
    float acc = ib0[w];
#pragma unroll
    for (int c = 0; c < CN; ++c) acc += ca[b * NI * CN + c] * iw0[w * CN + c];
    h0[idx] = fmaxf(acc, 0.f);
}
// init MLP layer 1: relu(h0 @ iw1.T + ib1)   [64,256]
__global__ void init_l1(const float* __restrict__ h0, const float* __restrict__ iw1,
                        const float* __restrict__ ib1, float* __restrict__ h1) {
    int idx = blockIdx.x * blockDim.x + threadIdx.x;
    if (idx >= BN * WN) return;
    int b = idx / WN, w = idx % WN;
    float acc = ib1[w];
    for (int k = 0; k < WN; ++k) acc += h0[b * WN + k] * iw1[w * WN + k];
    h1[idx] = fmaxf(acc, 0.f);
}
// init MLP layer 2: y0 = h1 @ iw2.T + ib2   [64,128]
__global__ void init_l2(const float* __restrict__ h1, const float* __restrict__ iw2,
                        const float* __restrict__ ib2, float* __restrict__ y0) {
    int idx = blockIdx.x * blockDim.x + threadIdx.x;
    if (idx >= BN * HN) return;
    int b = idx / HN, hh = idx % HN;
    float acc = ib2[hh];
    for (int k = 0; k < WN; ++k) acc += h1[b * WN + k] * iw2[hh * WN + k];
    y0[idx] = acc;
}

// readout: log_softmax(y @ lw.T + lb)
__global__ void readout_kernel(const float* __restrict__ y, const float* __restrict__ lw,
                               const float* __restrict__ lb, float* __restrict__ out) {
    int b = blockIdx.x * blockDim.x + threadIdx.x;
    if (b >= BN) return;
    float lg[OUTN];
    float mx = -3.4e38f;
#pragma unroll
    for (int o = 0; o < OUTN; ++o) {
        float acc = lb[o];
        for (int hh = 0; hh < HN; ++hh) acc += y[b * HN + hh] * lw[o * HN + hh];
        lg[o] = acc;
        mx = fmaxf(mx, acc);
    }
    float se = 0.f;
#pragma unroll
    for (int o = 0; o < OUTN; ++o) se += expf(lg[o] - mx);
    float lse = logf(se);
#pragma unroll
    for (int o = 0; o < OUTN; ++o) out[b * OUTN + o] = lg[o] - mx - lse;
}

// ---------------- main persistent ODE kernel ----------------
// grid = 4 workgroups (16 batch rows each, one per WGP), block = 256 (8 wave32).
// LDS: fw0 (64KB bf16) | fw1 (128KB bf16) | abf 16x256 bf16 | y 16x128 f32 |
//      ks[6] 16x128 f32 | dx 16x16 f32  => 263168 B (< 320KB/WGP)
__global__ __launch_bounds__(256, 1) void cde_main(
    const float* __restrict__ ts,
    const float* __restrict__ cd, const float* __restrict__ cc, const float* __restrict__ cb,
    float* __restrict__ y_ws,
    const __bf16* __restrict__ w01b,   // fw0b (256x128) immediately followed by fw1b (256x256)
    const __bf16* __restrict__ fw2b,   // 2048x256
    const float* __restrict__ fb0, const float* __restrict__ fb1, const float* __restrict__ fb2) {
    extern __shared__ char smem[];
    __bf16* s_fw0 = (__bf16*)smem;             // 256*128
    __bf16* s_fw1 = s_fw0 + WN * HN;           // 256*256
    __bf16* s_abf = s_fw1 + WN * WN;           // 16*256 (stage input / h1 / h2)
    float*  s_y   = (float*)(s_abf + 16 * WN); // 16*128
    float*  s_ks  = s_y + 16 * HN;             // 6*16*128
    float*  s_dx  = s_ks + 6 * 16 * HN;        // 16*16

    const int tid  = threadIdx.x;
    const int lane = tid & 31;
    const int wv   = tid >> 5;
    const int bg   = blockIdx.x;  // batch group: rows bg*16 .. bg*16+15

    // Stage fw0+fw1 (192KB bf16) into LDS with 16B vectors.
    {
        const uint4* src = (const uint4*)w01b;
        uint4* dst = (uint4*)smem;
        const int n16 = (WN * HN + WN * WN) * 2 / 16;  // 12288 * 16B
        for (int i = tid; i < n16; i += 256) dst[i] = src[i];
    }
    // Load this group's y0.
    for (int i = tid; i < 16 * HN; i += 256) s_y[i] = y_ws[bg * 16 * HN + i];
    __syncthreads();

    // Per-thread cubic-spline coefficients for dX/dt: thread owns (m = tid>>4, c = tid&15).
    const int m_dx = tid >> 4;
    const int c_dx = tid & 15;
    const int b_dx = bg * 16 + m_dx;

    const int lane_c  = lane & 15;        // column-in-tile / channel for this lane
    const int lane_mb = (lane >> 4) << 3; // row block base (0 or 8)

    for (int st = 0; st < NSTEP; ++st) {
        const int   iv   = st >> 1;
        const float hsub = (ts[iv + 1] - ts[iv]) * 0.5f;
        const float h    = hsub;
        const float fr0  = (st & 1) ? hsub : 0.f;

        // Hoist the per-interval spline coefficients once per step.
        const int   oc = (b_dx * NI + iv) * CN + c_dx;
        const float d3 = 3.f * cd[oc];
        const float c2 = 2.f * cc[oc];
        const float b1 = cb[oc];

        for (int s = 0; s < 6; ++s) {
            // --- stage input: ystage = y + h * sum_j a[s][j] * k_j  (bf16 into abf, stride HN)
            for (int i = tid; i < 16 * HN; i += 256) {
                float ys = s_y[i];
                for (int j = 0; j < s; ++j) ys += (h * TSA[s][j]) * s_ks[j * 16 * HN + i];
                s_abf[i] = f2bf(ys);
            }
            // --- dX/dt at this stage's frac
            {
                const float fr = fr0 + TSC[s] * h;
                s_dx[tid] = (d3 * fr + c2) * fr + b1;
            }
            __syncthreads();

            // ---------- GEMM1: h1[16,256] = softplus(ystage[16,128] @ fw0^T + fb0)
            // A fragments hoisted to registers; each feeds 2 WMMAs (2 N-tiles/wave).
            v8f acc0 = {0.f, 0.f, 0.f, 0.f, 0.f, 0.f, 0.f, 0.f};
            v8f acc1 = {0.f, 0.f, 0.f, 0.f, 0.f, 0.f, 0.f, 0.f};
            {
                const int nt0 = wv * 2, nt1 = wv * 2 + 1;
                v16bf afr[4];
#pragma unroll
                for (int kt = 0; kt < 4; ++kt) afr[kt] = load_frag_a(s_abf, kt, lane, HN);
#pragma unroll
                for (int kt = 0; kt < 4; ++kt) {
                    v16bf b0 = load_frag_b(s_fw0, nt0, kt, lane, HN);
                    v16bf b1 = load_frag_b(s_fw0, nt1, kt, lane, HN);
                    acc0 = __builtin_amdgcn_wmma_f32_16x16x32_bf16(false, afr[kt], false, b0,
                                                                   (short)0, acc0, false, false);
                    acc1 = __builtin_amdgcn_wmma_f32_16x16x32_bf16(false, afr[kt], false, b1,
                                                                   (short)0, acc1, false, false);
                }
            }
            __syncthreads();  // all waves done reading abf
            {
                const int n0 = (wv * 2) * 16 + lane_c;
                const int n1 = n0 + 16;
                const float bias0 = fb0[n0], bias1 = fb0[n1];
#pragma unroll
                for (int r = 0; r < 8; ++r) {
                    s_abf[(lane_mb + r) * WN + n0] = f2bf(softplusf(acc0[r] + bias0));
                    s_abf[(lane_mb + r) * WN + n1] = f2bf(softplusf(acc1[r] + bias1));
                }
            }
            __syncthreads();

            // ---------- GEMM2: h2[16,256] = softplus(h1[16,256] @ fw1^T + fb1)
            acc0 = (v8f){0.f, 0.f, 0.f, 0.f, 0.f, 0.f, 0.f, 0.f};
            acc1 = (v8f){0.f, 0.f, 0.f, 0.f, 0.f, 0.f, 0.f, 0.f};
            {
                const int nt0 = wv * 2, nt1 = wv * 2 + 1;
                v16bf afr[8];
#pragma unroll
                for (int kt = 0; kt < 8; ++kt) afr[kt] = load_frag_a(s_abf, kt, lane, WN);
#pragma unroll
                for (int kt = 0; kt < 8; ++kt) {
                    v16bf b0 = load_frag_b(s_fw1, nt0, kt, lane, WN);
                    v16bf b1 = load_frag_b(s_fw1, nt1, kt, lane, WN);
                    acc0 = __builtin_amdgcn_wmma_f32_16x16x32_bf16(false, afr[kt], false, b0,
                                                                   (short)0, acc0, false, false);
                    acc1 = __builtin_amdgcn_wmma_f32_16x16x32_bf16(false, afr[kt], false, b1,
                                                                   (short)0, acc1, false, false);
                }
            }
            __syncthreads();
            {
                const int n0 = (wv * 2) * 16 + lane_c;
                const int n1 = n0 + 16;
                const float bias0 = fb1[n0], bias1 = fb1[n1];
#pragma unroll
                for (int r = 0; r < 8; ++r) {
                    s_abf[(lane_mb + r) * WN + n0] = f2bf(softplusf(acc0[r] + bias0));
                    s_abf[(lane_mb + r) * WN + n1] = f2bf(softplusf(acc1[r] + bias1));
                }
            }
            __syncthreads();

            // ---------- GEMM3 + contraction:
            // v[16,2048] = tanh(h2 @ fw2^T + fb2); k_s[m,hid] = sum_c v[m, hid*16+c]*dx[m,c].
            // N-tile nt == hidden unit index; tile columns == the 16 channels.
            // A (whole h2 row tile, K=256) lives in 8 register fragments for all 16 tiles;
            // dx values are loop-invariant per lane -> registers too.
            {
                float* ksd = s_ks + s * 16 * HN;
                v16bf afr[8];
#pragma unroll
                for (int kt = 0; kt < 8; ++kt) afr[kt] = load_frag_a(s_abf, kt, lane, WN);
                float dxr[8];
#pragma unroll
                for (int r = 0; r < 8; ++r) dxr[r] = s_dx[(lane_mb + r) * 16 + lane_c];

                for (int t = 0; t < 8; ++t) {  // two hidden tiles per iteration
                    const int nt0 = wv * 16 + 2 * t;
                    const int nt1 = nt0 + 1;
                    if (t < 7) {  // prefetch next pair's B rows into L2/L0
                        __builtin_prefetch(fw2b + (nt0 + 2) * 16 * WN + lane * 128, 0, 1);
                        __builtin_prefetch(fw2b + (nt1 + 2) * 16 * WN + lane * 128, 0, 1);
                    }
                    v8f a0 = {0.f, 0.f, 0.f, 0.f, 0.f, 0.f, 0.f, 0.f};
                    v8f a1 = {0.f, 0.f, 0.f, 0.f, 0.f, 0.f, 0.f, 0.f};
#pragma unroll
                    for (int kt = 0; kt < 8; ++kt) {
                        v16bf b0 = load_frag_b(fw2b, nt0, kt, lane, WN);
                        v16bf b1 = load_frag_b(fw2b, nt1, kt, lane, WN);
                        a0 = __builtin_amdgcn_wmma_f32_16x16x32_bf16(false, afr[kt], false, b0,
                                                                     (short)0, a0, false, false);
                        a1 = __builtin_amdgcn_wmma_f32_16x16x32_bf16(false, afr[kt], false, b1,
                                                                     (short)0, a1, false, false);
                    }
                    const float bias0 = fb2[nt0 * 16 + lane_c];
                    const float bias1 = fb2[nt1 * 16 + lane_c];
#pragma unroll
                    for (int r = 0; r < 8; ++r) {
                        float p0 = tanhf(a0[r] + bias0) * dxr[r];
                        float p1 = tanhf(a1[r] + bias1) * dxr[r];
                        p0 += __shfl_xor(p0, 1);
                        p1 += __shfl_xor(p1, 1);
                        p0 += __shfl_xor(p0, 2);
                        p1 += __shfl_xor(p1, 2);
                        p0 += __shfl_xor(p0, 4);
                        p1 += __shfl_xor(p1, 4);
                        p0 += __shfl_xor(p0, 8);
                        p1 += __shfl_xor(p1, 8);
                        if (lane_c == 0) {
                            ksd[(lane_mb + r) * HN + nt0] = p0;
                            ksd[(lane_mb + r) * HN + nt1] = p1;
                        }
                    }
                }
            }
            __syncthreads();
        }  // stages

        // y += h * sum_j b[j] * k_j
        for (int i = tid; i < 16 * HN; i += 256) {
            float yv = s_y[i];
#pragma unroll
            for (int j = 0; j < 6; ++j) yv += (h * TSB[j]) * s_ks[j * 16 * HN + i];
            s_y[i] = yv;
        }
        __syncthreads();
    }  // steps

    for (int i = tid; i < 16 * HN; i += 256) y_ws[bg * 16 * HN + i] = s_y[i];
}

// ---------------- launch ----------------

extern "C" void kernel_launch(void* const* d_in, const int* in_sizes, int n_in,
                              void* d_out, int out_size, void* d_ws, size_t ws_size,
                              hipStream_t stream) {
    const float* ts  = (const float*)d_in[0];
    const float* cd  = (const float*)d_in[1];
    const float* cc  = (const float*)d_in[2];
    const float* cb  = (const float*)d_in[3];
    const float* ca  = (const float*)d_in[4];
    const float* iw0 = (const float*)d_in[5];
    const float* ib0 = (const float*)d_in[6];
    const float* iw1 = (const float*)d_in[7];
    const float* ib1 = (const float*)d_in[8];
    const float* iw2 = (const float*)d_in[9];
    const float* ib2 = (const float*)d_in[10];
    const float* fw0 = (const float*)d_in[11];
    const float* fb0 = (const float*)d_in[12];
    const float* fw1 = (const float*)d_in[13];
    const float* fb1 = (const float*)d_in[14];
    const float* fw2 = (const float*)d_in[15];
    const float* fb2 = (const float*)d_in[16];
    const float* lw  = (const float*)d_in[17];
    const float* lb  = (const float*)d_in[18];

    char* ws = (char*)d_ws;
    float* y_ws  = (float*)(ws + 0);        // 64*128*4   = 32768
    float* h0_ws = (float*)(ws + 32768);    // 64*256*4   = 65536
    float* h1_ws = (float*)(ws + 98304);    // 64*256*4   = 65536
    __hip_bfloat16* fw0b = (__hip_bfloat16*)(ws + 163840);  // 256*128*2 = 65536
    __hip_bfloat16* fw1b = (__hip_bfloat16*)(ws + 229376);  // 256*256*2 = 131072
    __hip_bfloat16* fw2b = (__hip_bfloat16*)(ws + 360448);  // 2048*256*2 = 1048576

    // Weight conversion fp32 -> bf16 (row-major layout == WMMA B-fragment friendly).
    to_bf16_kernel<<<(WN * HN + 255) / 256, 256, 0, stream>>>(fw0, fw0b, WN * HN);
    to_bf16_kernel<<<(WN * WN + 255) / 256, 256, 0, stream>>>(fw1, fw1b, WN * WN);
    to_bf16_kernel<<<(HN * CN * WN + 255) / 256, 256, 0, stream>>>(fw2, fw2b, HN * CN * WN);

    // Initial MLP: y0 = init_mlp(coeff_a[:,0]).
    init_l0<<<(BN * WN + 255) / 256, 256, 0, stream>>>(ca, iw0, ib0, h0_ws);
    init_l1<<<(BN * WN + 255) / 256, 256, 0, stream>>>(h0_ws, iw1, ib1, h1_ws);
    init_l2<<<(BN * HN + 255) / 256, 256, 0, stream>>>(h1_ws, iw2, ib2, y_ws);

    // Main ODE integration: 4 workgroups x 256 threads, ~257KB dynamic LDS each.
    const size_t smem = (size_t)(WN * HN + WN * WN + 16 * WN) * 2 +
                        (size_t)(16 * HN + 6 * 16 * HN + 16 * 16) * 4;
    cde_main<<<4, 256, smem, stream>>>(ts, cd, cc, cb, y_ws,
                                       (const __bf16*)fw0b, (const __bf16*)fw2b,
                                       fb0, fb1, fb2);

    // Readout + log_softmax.
    readout_kernel<<<1, 64, 0, stream>>>(y_ws, lw, lb, (float*)d_out);
}
